// PeepholeLSTMCell_65240553226497
// MI455X (gfx1250) — matrix operand order repaired and checked by reading, not compiled, not run on hardware
//
#include <hip/hip_runtime.h>
#include <hip/hip_bf16.h>

#define B_DIM 4096
#define I_DIM 1024
#define H_DIM 1024
#define K_DIM 2048  // I + H

typedef __bf16 bf16x8  __attribute__((ext_vector_type(8)));
typedef __bf16 bf16x16 __attribute__((ext_vector_type(16)));
typedef float  f32x8   __attribute__((ext_vector_type(8)));
typedef float  f32x4   __attribute__((ext_vector_type(4)));

__device__ __forceinline__ float fast_sigmoid(float x) {
    return 1.0f / (1.0f + __expf(-x));
}
__device__ __forceinline__ float fast_tanh(float x) {
    return 1.0f - 2.0f / (__expf(2.0f * x) + 1.0f);
}
__device__ __forceinline__ bf16x16 join16(bf16x8 lo, bf16x8 hi) {
    return __builtin_shufflevector(lo, hi, 0, 1, 2, 3, 4, 5, 6, 7,
                                            8, 9, 10, 11, 12, 13, 14, 15);
}
__device__ __forceinline__ bf16x8 cvt8(f32x4 a, f32x4 b) {
    bf16x8 o;
    o[0] = (__bf16)a[0]; o[1] = (__bf16)a[1]; o[2] = (__bf16)a[2]; o[3] = (__bf16)a[3];
    o[4] = (__bf16)b[0]; o[5] = (__bf16)b[1]; o[6] = (__bf16)b[2]; o[7] = (__bf16)b[3];
    return o;
}
// A fragment: lane holds 16 contiguous K = two adjacent b128 chunks
__device__ __forceinline__ bf16x16 loadA(const __bf16* p) {
    return join16(*(const bf16x8*)(p), *(const bf16x8*)(p + 8));
}
// B fragment: lane holds 16 contiguous K = two adjacent b128 chunks
__device__ __forceinline__ bf16x16 loadB(const __bf16* p) {
    return join16(*(const bf16x8*)(p), *(const bf16x8*)(p + 8));
}
__device__ __forceinline__ f32x8 wmma_bf16(bf16x16 a, bf16x16 b, f32x8 c) {
    return __builtin_amdgcn_wmma_f32_16x16x32_bf16(false, a, false, b,
                                                   (short)0, c, false, false);
}

// ---- Pack x||h (fp32) -> A bf16 [B_DIM][K_DIM], K contiguous ----
__global__ void pack_A_kernel(const float* __restrict__ x,
                              const float* __restrict__ h,
                              __bf16* __restrict__ A) {
    const int tid = blockIdx.x * blockDim.x + threadIdx.x;  // one 8-elem chunk
    const int row = tid >> 8;                               // 256 chunks / row
    const int k   = (tid & 255) << 3;
    const float* src = (k < I_DIM) ? (x + (size_t)row * I_DIM + k)
                                   : (h + (size_t)row * H_DIM + (k - I_DIM));
    f32x4 v0 = *(const f32x4*)(src);
    f32x4 v1 = *(const f32x4*)(src + 4);
    *(bf16x8*)(A + (size_t)row * K_DIM + k) = cvt8(v0, v1);
}

// ---- Pack 8 weight mats (fp32) -> W bf16 [4 gates][H_DIM rows][K_DIM] ----
__global__ void pack_W_kernel(const float* __restrict__ wii, const float* __restrict__ wif,
                              const float* __restrict__ wig, const float* __restrict__ wio,
                              const float* __restrict__ whi, const float* __restrict__ whf,
                              const float* __restrict__ whg, const float* __restrict__ who,
                              __bf16* __restrict__ W) {
    const int tid = blockIdx.x * blockDim.x + threadIdx.x;  // one 8-elem chunk
    const int g   = tid >> 18;              // 1024 rows * 256 chunks = 2^18 per gate
    const int rem = tid & 262143;
    const int n   = rem >> 8;
    const int k   = (rem & 255) << 3;
    const float* gx = (g == 0) ? wii : (g == 1) ? wif : (g == 2) ? wig : wio;
    const float* gh = (g == 0) ? whi : (g == 1) ? whf : (g == 2) ? whg : who;
    const float* src = (k < I_DIM) ? (gx + (size_t)n * I_DIM + k)
                                   : (gh + (size_t)n * H_DIM + (k - I_DIM));
    f32x4 v0 = *(const f32x4*)(src);
    f32x4 v1 = *(const f32x4*)(src + 4);
    *(bf16x8*)(W + ((size_t)g * H_DIM + n) * K_DIM + k) = cvt8(v0, v1);
}

// 8 independent WMMAs: 2 M-subtiles x 4 gates on one fragment set
#define WMMA_GROUP(A0, A1, B0, B1, B2, B3)          \
    do {                                            \
        acc[0][0] = wmma_bf16(A0, B0, acc[0][0]);   \
        acc[0][1] = wmma_bf16(A1, B0, acc[0][1]);   \
        acc[1][0] = wmma_bf16(A0, B1, acc[1][0]);   \
        acc[1][1] = wmma_bf16(A1, B1, acc[1][1]);   \
        acc[2][0] = wmma_bf16(A0, B2, acc[2][0]);   \
        acc[2][1] = wmma_bf16(A1, B2, acc[2][1]);   \
        acc[3][0] = wmma_bf16(A0, B3, acc[3][0]);   \
        acc[3][1] = wmma_bf16(A1, B3, acc[3][1]);   \
    } while (0)

// ---- Fused GEMM (bf16 WMMA, f32 accum) + peephole LSTM epilogue ----
// grid: (H_DIM/32, B_DIM/128), block: 256 threads = 8 wave32s (4 M x 2 N)
// Each wave: 32 M-rows x 16 N-cols x 4 gates -> 8 accumulators.
// Ping-pong (P/Q) fragment buffers, unroll-by-2 pipeline: no register rotation.
__global__ __launch_bounds__(256)
void lstm_wmma_kernel(const __bf16* __restrict__ A,
                      const __bf16* __restrict__ W,
                      const float* __restrict__ cprev,
                      const float* __restrict__ bii, const float* __restrict__ bhi,
                      const float* __restrict__ bif, const float* __restrict__ bhf,
                      const float* __restrict__ big, const float* __restrict__ bhg,
                      const float* __restrict__ bio, const float* __restrict__ bho,
                      const float* __restrict__ Wci, const float* __restrict__ Wcf,
                      const float* __restrict__ Wco,
                      float* __restrict__ out) {
    const int lane = threadIdx.x & 31;
    const int wave = threadIdx.x >> 5;
    const int wm = wave & 3;   // 0..3 along M (batch), 32 rows each
    const int wn = wave >> 2;  // 0..1 along N (hidden cols), 16 each
    const int Mbase = blockIdx.y * 128 + wm * 32;
    const int Nbase = blockIdx.x * 32 + wn * 16;

    const int kh = lane >> 4;
    // Two A row-tiles per wave
    const __bf16* ap0 = A + (size_t)(Mbase      + (lane & 15)) * K_DIM + kh * 16;
    const __bf16* ap1 = A + (size_t)(Mbase + 16 + (lane & 15)) * K_DIM + kh * 16;
    // Four gate B tiles
    const int hcol = Nbase + (lane & 15);
    const __bf16* wp0 = W + ((size_t)0 * H_DIM + hcol) * K_DIM + kh * 16;
    const __bf16* wp1 = W + ((size_t)1 * H_DIM + hcol) * K_DIM + kh * 16;
    const __bf16* wp2 = W + ((size_t)2 * H_DIM + hcol) * K_DIM + kh * 16;
    const __bf16* wp3 = W + ((size_t)3 * H_DIM + hcol) * K_DIM + kh * 16;

    f32x8 acc[4][2];
#pragma unroll
    for (int g = 0; g < 4; ++g) {
        acc[g][0] = (f32x8){};
        acc[g][1] = (f32x8){};
    }

    // Prologue: P holds kk=0, Q holds kk=32
    bf16x16 pa0 = loadA(ap0);
    bf16x16 pa1 = loadA(ap1);
    bf16x16 pb0 = loadB(wp0);
    bf16x16 pb1 = loadB(wp1);
    bf16x16 pb2 = loadB(wp2);
    bf16x16 pb3 = loadB(wp3);
    bf16x16 qa0 = loadA(ap0 + 32);
    bf16x16 qa1 = loadA(ap1 + 32);
    bf16x16 qb0 = loadB(wp0 + 32);
    bf16x16 qb1 = loadB(wp1 + 32);
    bf16x16 qb2 = loadB(wp2 + 32);
    bf16x16 qb3 = loadB(wp3 + 32);

#pragma unroll 1
    for (int kk = 64; kk < K_DIM; kk += 64) {
        // Consume P, refill P in place with fragments at kk
        WMMA_GROUP(pa0, pa1, pb0, pb1, pb2, pb3);
        pa0 = loadA(ap0 + kk);
        pa1 = loadA(ap1 + kk);
        pb0 = loadB(wp0 + kk);
        pb1 = loadB(wp1 + kk);
        pb2 = loadB(wp2 + kk);
        pb3 = loadB(wp3 + kk);
        // Consume Q, refill Q in place with fragments at kk+32
        WMMA_GROUP(qa0, qa1, qb0, qb1, qb2, qb3);
        qa0 = loadA(ap0 + kk + 32);
        qa1 = loadA(ap1 + kk + 32);
        qb0 = loadB(wp0 + kk + 32);
        qb1 = loadB(wp1 + kk + 32);
        qb2 = loadB(wp2 + kk + 32);
        qb3 = loadB(wp3 + kk + 32);
    }
    // Tail: final two K-steps
    WMMA_GROUP(pa0, pa1, pb0, pb1, pb2, pb3);
    WMMA_GROUP(qa0, qa1, qb0, qb1, qb2, qb3);

    // LSTM epilogue. C/D layout: lane&15 = N col, VGPR j = row (+8 if lane>=16)
    const float bsi = bii[hcol] + bhi[hcol];
    const float bsf = bif[hcol] + bhf[hcol];
    const float bsg = big[hcol] + bhg[hcol];
    const float bso = bio[hcol] + bho[hcol];
    const float wci = Wci[hcol];
    const float wcf = Wcf[hcol];
    const float wco = Wco[hcol];
    float* hout = out;
    float* cout = out + (size_t)B_DIM * H_DIM;

#pragma unroll
    for (int s = 0; s < 2; ++s) {
        const int mrow0 = Mbase + s * 16 + (lane >> 4) * 8;
#pragma unroll
        for (int j = 0; j < 8; ++j) {
            const int brow = mrow0 + j;
            const float cv = cprev[(size_t)brow * H_DIM + hcol];
            const float ig = fast_sigmoid(acc[0][s][j] + bsi + wci * cv);
            const float fg = fast_sigmoid(acc[1][s][j] + bsf + wcf * cv);
            const float gg = fast_tanh(acc[2][s][j] + bsg);
            const float cn = fg * cv + ig * gg;
            const float og = fast_sigmoid(acc[3][s][j] + bso + wco * cn);
            const float hn = og * fast_tanh(cn);
            hout[(size_t)brow * H_DIM + hcol] = hn;
            cout[(size_t)brow * H_DIM + hcol] = cn;
        }
    }
}

extern "C" void kernel_launch(void* const* d_in, const int* in_sizes, int n_in,
                              void* d_out, int out_size, void* d_ws, size_t ws_size,
                              hipStream_t stream) {
    const float* x    = (const float*)d_in[0];
    const float* h    = (const float*)d_in[1];
    const float* c    = (const float*)d_in[2];
    const float* Wii  = (const float*)d_in[3];
    const float* bii  = (const float*)d_in[4];
    const float* Whi  = (const float*)d_in[5];
    const float* bhi  = (const float*)d_in[6];
    const float* Wci  = (const float*)d_in[7];
    const float* Wif  = (const float*)d_in[8];
    const float* bif  = (const float*)d_in[9];
    const float* Whf  = (const float*)d_in[10];
    const float* bhf  = (const float*)d_in[11];
    const float* Wcf  = (const float*)d_in[12];
    const float* Wig  = (const float*)d_in[13];
    const float* big  = (const float*)d_in[14];
    const float* Whg  = (const float*)d_in[15];
    const float* bhg  = (const float*)d_in[16];
    const float* Wio  = (const float*)d_in[17];
    const float* bio  = (const float*)d_in[18];
    const float* Who  = (const float*)d_in[19];
    const float* bho  = (const float*)d_in[20];
    const float* Wco  = (const float*)d_in[21];

    __bf16* Abf = (__bf16*)d_ws;                         // 16 MiB
    __bf16* Wbf = Abf + (size_t)B_DIM * K_DIM;           // 16 MiB

    // 4096*2048/8 = 1M chunks -> 4096 blocks of 256
    pack_A_kernel<<<4096, 256, 0, stream>>>(x, h, Abf);
    // 4*1024*2048/8 = 1M chunks -> 4096 blocks of 256
    pack_W_kernel<<<4096, 256, 0, stream>>>(Wii, Wif, Wig, Wio,
                                            Whi, Whf, Whg, Who, Wbf);

    dim3 grid(H_DIM / 32, B_DIM / 128);  // (32, 32)
    lstm_wmma_kernel<<<grid, 256, 0, stream>>>(Abf, Wbf, c,
                                               bii, bhi, bif, bhf,
                                               big, bhg, bio, bho,
                                               Wci, Wcf, Wco,
                                               (float*)d_out);
}
